// Encoder_76166950028615
// MI455X (gfx1250) — matrix operand order repaired and checked by reading, not compiled, not run on hardware
//
#include <hip/hip_runtime.h>

#define DM 2048
#define HEADS 16
#define HK 128
#define FFN_DIM 8192
#define SEQ 2048
#define BS 4096  // B * S rows

typedef _Float16 v16h __attribute__((ext_vector_type(16)));
typedef _Float16 v8h __attribute__((ext_vector_type(8)));
typedef float v8f __attribute__((ext_vector_type(8)));
typedef int v4i __attribute__((ext_vector_type(4)));  // native vector (builtin param type)

#if defined(__HIP_DEVICE_COMPILE__) && __has_builtin(__builtin_amdgcn_global_load_async_to_lds_b128)
#define ASYNC_CP 1
#else
#define ASYNC_CP 0
#endif

__device__ __forceinline__ void cp_b128(_Float16* l, const _Float16* g) {
#if ASYNC_CP
  __builtin_amdgcn_global_load_async_to_lds_b128(
      (v4i*)const_cast<_Float16*>(g), (v4i*)l, 0, 0);
#else
  *(v4i*)l = *(const v4i*)g;
#endif
}

__device__ __forceinline__ void cp_wait() {
#if ASYNC_CP
#if __has_builtin(__builtin_amdgcn_s_wait_asynccnt)
  __builtin_amdgcn_s_wait_asynccnt(0);
#else
  asm volatile("s_wait_asynccnt 0" ::: "memory");
#endif
#endif
}

__device__ __forceinline__ v8f wmma_f16(v16h a, v16h b, v8f c) {
  // D = A(16x32) * B(32x16) + C, fp32 accumulate
  return __builtin_amdgcn_wmma_f32_16x16x32_f16(false, a, false, b, (short)0, c,
                                                false, false);
}

// ---- fragment loaders (ISA 7.12.2 layouts), all 2x ds_load_b128 per lane ----
// A 16x32 f16: lane<16 -> m=lane, k={0..7,16..23}; lane>=16 -> m=lane-16, k={8..15,24..31}
__device__ __forceinline__ v16h frag_a(const _Float16* p, int row0, int ld) {
  const int lane = threadIdx.x & 31;
  const int m = lane & 15;
  const int kb = (lane & 16) ? 8 : 0;
  union { v16h v; v8h q[2]; } u;
  const _Float16* r = p + (size_t)(row0 + m) * ld + kb;
  u.q[0] = *(const v8h*)r;
  u.q[1] = *(const v8h*)(r + 16);
  return u.v;
}

// B 32x16 f16 from K-contiguous (n-major) storage: B[k][n] = p[(n0+n)*ld + k]
// lane<16 -> n=lane, k={0..7,16..23}; lane>=16 -> n=lane-16, k={8..15,24..31}
__device__ __forceinline__ v16h frag_b_kc(const _Float16* p, int n0, int ld) {
  const int lane = threadIdx.x & 31;
  const int n = lane & 15;
  const int kb = (lane & 16) ? 8 : 0;
  union { v16h v; v8h q[2]; } u;
  const _Float16* r = p + (size_t)(n0 + n) * ld + kb;
  u.q[0] = *(const v8h*)r;
  u.q[1] = *(const v8h*)(r + 16);
  return u.v;
}

// ---- conversion kernels (weights are pre-transposed so GEMM B is K-contig) ----
__global__ void cvt_f32_f16(const float* __restrict__ s, _Float16* __restrict__ d, int n) {
  int i = blockIdx.x * 256 + threadIdx.x;
  if (i < n) d[i] = (_Float16)s[i];
}

__global__ void cvt_transpose_f16(const float* __restrict__ s, _Float16* __restrict__ d,
                                  int R, int C) {
  int i = blockIdx.x * 256 + threadIdx.x;
  if (i < R * C) {
    int r = i / C, c = i - r * C;
    d[(size_t)c * R + r] = (_Float16)s[i];  // d is [C, R]
  }
}

// ---- RMSNorm (no eps, matches reference) fp32 in -> f16 out ----
__global__ __launch_bounds__(256) void rmsnorm_kernel(const float* __restrict__ x,
                                                      _Float16* __restrict__ y, int cols) {
  const int row = blockIdx.x;
  const float* xr = x + (size_t)row * cols;
  float s = 0.f;
  for (int i = threadIdx.x; i < cols; i += 256) {
    float v = xr[i];
    s += v * v;
  }
#pragma unroll
  for (int off = 16; off; off >>= 1) s += __shfl_xor(s, off, 32);
  __shared__ float red[8];
  __shared__ float sc;
  if ((threadIdx.x & 31) == 0) red[threadIdx.x >> 5] = s;
  __syncthreads();
  if (threadIdx.x == 0) {
    float t = 0.f;
#pragma unroll
    for (int i = 0; i < 8; ++i) t += red[i];
    sc = rsqrtf(t / (float)cols);
  }
  __syncthreads();
  const float k = sc;
  for (int i = threadIdx.x; i < cols; i += 256)
    y[(size_t)row * cols + i] = (_Float16)(xr[i] * k);
}

// ---- WMMA GEMM: C[M,N] = A[M,K] * Bt^T, Bt given N-major [N,K] ----
// block tile 64x128, 4 waves (2x2) each 32x64 (2x4 frags), BK=32
// EPI: 0 f16 out; 1 bias+ReLU6 f16; 2 +residual f32; 3 bias+residual f32;
//      4 f16 out in V-transposed layout [b,h,d,s]
template <int EPI>
__global__ __launch_bounds__(128) void gemm_f16w(
    const _Float16* __restrict__ A, const _Float16* __restrict__ Bt, int M, int N, int Kd,
    const float* __restrict__ bias, const float* __restrict__ resid,
    _Float16* __restrict__ outH, float* __restrict__ outF) {
  __shared__ __align__(16) _Float16 As[64 * 32];    // row-major, K contig
  __shared__ __align__(16) _Float16 Bs[128 * 32];   // n-major, K contig
  const int tid = threadIdx.x;
  const int wave = tid >> 5;
  const int lane = tid & 31;
  const int rowBase = blockIdx.y * 64;
  const int colBase = blockIdx.x * 128;
  const int wr = (wave >> 1) * 32;
  const int wc = (wave & 1) * 64;
  v8f acc[2][4] = {};

  for (int k0 = 0; k0 < Kd; k0 += 32) {
    __syncthreads();  // previous iteration's fragment reads complete
    // A tile 64x32 -> 256 16B vectors
    for (int v = tid; v < 256; v += 128) {
      int r = v >> 2, cc = (v & 3) << 3;
      const _Float16* g = &A[(size_t)(rowBase + r) * Kd + k0 + cc];
      cp_b128(&As[r * 32 + cc], g);
      if (k0 + 32 < Kd) __builtin_prefetch(g + 32, 0, 1);
    }
    // B tile 128x32 (n-major) -> 512 16B vectors
    for (int v = tid; v < 512; v += 128) {
      int r = v >> 2, cc = (v & 3) << 3;
      const _Float16* g = &Bt[(size_t)(colBase + r) * Kd + k0 + cc];
      cp_b128(&Bs[r * 32 + cc], g);
      if (k0 + 32 < Kd) __builtin_prefetch(g + 32, 0, 1);
    }
    cp_wait();
    __syncthreads();
    v16h a0 = frag_a(As, wr + 0, 32);
    v16h a1 = frag_a(As, wr + 16, 32);
#pragma unroll
    for (int f = 0; f < 4; ++f) {
      v16h b = frag_b_kc(Bs, wc + f * 16, 32);
      acc[0][f] = wmma_f16(a0, b, acc[0][f]);
      acc[1][f] = wmma_f16(a1, b, acc[1][f]);
    }
  }

  const int nl = lane & 15;
  const int mh = (lane & 16) ? 8 : 0;
#pragma unroll
  for (int fr = 0; fr < 2; ++fr)
#pragma unroll
    for (int fc = 0; fc < 4; ++fc)
#pragma unroll
      for (int r = 0; r < 8; ++r) {
        int row = rowBase + wr + fr * 16 + r + mh;
        int col = colBase + wc + fc * 16 + nl;
        size_t oi = (size_t)row * N + col;
        float vv = acc[fr][fc][r];
        if (EPI == 0) {
          outH[oi] = (_Float16)vv;
        } else if (EPI == 1) {
          vv += bias[col];
          vv = fminf(fmaxf(vv, 0.f), 6.f);
          outH[oi] = (_Float16)vv;
        } else if (EPI == 2) {
          outF[oi] = vv + resid[oi];
        } else if (EPI == 3) {
          outF[oi] = vv + bias[col] + resid[oi];
        } else {
          // V transposed: [b, h, d, s]; (b*16+h)*128+d == b*2048+col
          size_t ti = ((size_t)((row >> 11) * DM + col)) * SEQ + (row & (SEQ - 1));
          outH[ti] = (_Float16)vv;
        }
      }
}

// ---- causal flash attention, WMMA for QK^T and PV ----
// grid: (SEQ/64, HEADS, B); block 128; wave w handles q rows [q0, q0+16)
__global__ __launch_bounds__(128) void attn_kernel(const _Float16* __restrict__ qb,
                                                   const _Float16* __restrict__ kb,
                                                   const _Float16* __restrict__ vT,
                                                   _Float16* __restrict__ ob) {
  __shared__ __align__(16) _Float16 Kt[32 * 128];   // [kv, d] row-major
  __shared__ __align__(16) _Float16 Vt[128 * 32];   // [d, kv] -> K-contig for PV B-frags
  __shared__ __align__(16) _Float16 Ps[4][16 * 32];
  const int tid = threadIdx.x, wave = tid >> 5, lane = tid & 31;
  const int bb = blockIdx.z, hh = blockIdx.y;
  const int q0 = blockIdx.x * 64 + wave * 16;
  const size_t rowQ = (size_t)bb * SEQ + q0;
  const int nl = lane & 15;
  const bool hi = (lane & 16) != 0;

  // load Q fragments once (16 rows x 128 head dims = 4 A-frags), 2x b128 each
  v16h qf[4];
  {
    const _Float16* qrow = qb + (rowQ + nl) * DM + hh * HK + (hi ? 8 : 0);
#pragma unroll
    for (int c = 0; c < 4; ++c) {
      union { v16h v; v8h q[2]; } u;
      u.q[0] = *(const v8h*)(qrow + c * 32);
      u.q[1] = *(const v8h*)(qrow + c * 32 + 16);
      qf[c] = u.v;
    }
  }

  float mrow[8], lrow[8];
  v8f oacc[8] = {};
#pragma unroll
  for (int r = 0; r < 8; ++r) {
    mrow[r] = -3.0e38f;
    lrow[r] = 0.f;
  }
  const float scale = 0.08838834764831845f;  // 1/sqrt(128)
  const int kvEnd = blockIdx.x * 64 + 64;    // uniform causal bound for the block

  for (int kv0 = 0; kv0 < kvEnd; kv0 += 32) {
    __syncthreads();
    // K tile: 32 kv-rows x 128 d (row-major)
    for (int v = tid; v < 512; v += 128) {
      int r = v >> 4, cc = (v & 15) << 3;
      cp_b128(&Kt[r * 128 + cc], &kb[((size_t)bb * SEQ + kv0 + r) * DM + hh * HK + cc]);
    }
    // V tile from transposed V: 128 d-rows x 32 kv (K-contig)
    for (int v = tid; v < 512; v += 128) {
      int d = v >> 2, cc = (v & 3) << 3;
      cp_b128(&Vt[d * 32 + cc],
              &vT[((size_t)(bb * DM + hh * HK + d)) * SEQ + kv0 + cc]);
    }
    cp_wait();
    __syncthreads();

    // S = Q * K^T for 32 kv positions (2 N-frags x 4 k-chunks)
    v8f s0 = {}, s1 = {};
#pragma unroll
    for (int c = 0; c < 4; ++c) {
      v16h bk0 = frag_b_kc(Kt + c * 32, 0, 128);
      v16h bk1 = frag_b_kc(Kt + c * 32, 16, 128);
      s0 = wmma_f16(qf[c], bk0, s0);
      s1 = wmma_f16(qf[c], bk1, s1);
    }

    // online softmax update (rows live in 16-lane groups)
    _Float16* pp = &Ps[wave][0];
#pragma unroll
    for (int r = 0; r < 8; ++r) {
      const int qg = q0 + r + (hi ? 8 : 0);
      float c0 = s0[r] * scale;
      if (kv0 + nl > qg) c0 = -3.0e38f;
      float c1 = s1[r] * scale;
      if (kv0 + 16 + nl > qg) c1 = -3.0e38f;
      float cm = fmaxf(c0, c1);
#pragma unroll
      for (int off = 8; off; off >>= 1) cm = fmaxf(cm, __shfl_xor(cm, off, 32));
      const float mnew = fmaxf(mrow[r], cm);
      const float alpha = __expf(mrow[r] - mnew);
      const float p0 = __expf(c0 - mnew);
      const float p1 = __expf(c1 - mnew);
      float ps = p0 + p1;
#pragma unroll
      for (int off = 8; off; off >>= 1) ps += __shfl_xor(ps, off, 32);
      mrow[r] = mnew;
      lrow[r] = lrow[r] * alpha + ps;
#pragma unroll
      for (int f = 0; f < 8; ++f) oacc[f][r] *= alpha;
      const int mm = r + (hi ? 8 : 0);
      pp[mm * 32 + nl] = (_Float16)p0;
      pp[mm * 32 + 16 + nl] = (_Float16)p1;
    }
    // wave-local LDS handoff: P stores must land before A-frag reload
    asm volatile("s_wait_dscnt 0" ::: "memory");

    // O += P(16x32) * V(32x128): 8 WMMAs, all-b128 fragment loads
    v16h pa = frag_a(pp, 0, 32);
#pragma unroll
    for (int f = 0; f < 8; ++f) {
      v16h bv = frag_b_kc(Vt, f * 16, 32);
      oacc[f] = wmma_f16(pa, bv, oacc[f]);
    }
  }

  // normalize and write f16 output [BS, H*K]
#pragma unroll
  for (int f = 0; f < 8; ++f) {
#pragma unroll
    for (int r = 0; r < 8; ++r) {
      const int m = r + (hi ? 8 : 0);
      float vv = oacc[f][r] / lrow[r];
      ob[(rowQ + m) * DM + hh * HK + f * 16 + nl] = (_Float16)vv;
    }
  }
}

extern "C" void kernel_launch(void* const* d_in, const int* in_sizes, int n_in, void* d_out,
                              int out_size, void* d_ws, size_t ws_size, hipStream_t stream) {
  (void)in_sizes; (void)n_in; (void)out_size; (void)ws_size;
  const float* x = (const float*)d_in[0];
  const float* wq = (const float*)d_in[1];
  const float* wk = (const float*)d_in[2];
  const float* wv = (const float*)d_in[3];
  const float* wo = (const float*)d_in[4];
  const float* w1 = (const float*)d_in[5];
  const float* b1 = (const float*)d_in[6];
  const float* w2 = (const float*)d_in[7];
  const float* b2 = (const float*)d_in[8];
  float* out = (float*)d_out;

  char* ws = (char*)d_ws;
  size_t off = 0;
  auto alloc = [&](size_t bytes) -> char* {
    char* p = ws + off;
    off += (bytes + 255) & ~(size_t)255;
    return p;
  };
  _Float16* xn   = (_Float16*)alloc((size_t)BS * DM * 2);       // normed acts (reused)
  _Float16* qb   = (_Float16*)alloc((size_t)BS * DM * 2);
  _Float16* kbuf = (_Float16*)alloc((size_t)BS * DM * 2);
  _Float16* vbufT= (_Float16*)alloc((size_t)BS * DM * 2);       // [b,h,d,s]
  _Float16* attn = (_Float16*)alloc((size_t)BS * DM * 2);
  _Float16* hbuf = (_Float16*)alloc((size_t)BS * FFN_DIM * 2);
  _Float16* wqT  = (_Float16*)alloc((size_t)DM * DM * 2);       // [N=HK, K=E]
  _Float16* wkT  = (_Float16*)alloc((size_t)DM * DM * 2);
  _Float16* wvT  = (_Float16*)alloc((size_t)DM * DM * 2);
  _Float16* woB  = (_Float16*)alloc((size_t)DM * DM * 2);       // w_o as-is: [N=E, K=HK]
  _Float16* w1T  = (_Float16*)alloc((size_t)DM * FFN_DIM * 2);  // [FFN, D]
  _Float16* w2T  = (_Float16*)alloc((size_t)FFN_DIM * DM * 2);  // [D, FFN]
  float* x2      = (float*)alloc((size_t)BS * DM * 4);          // post-attn residual

  const int nW = DM * DM;
  cvt_transpose_f16<<<(nW + 255) / 256, 256, 0, stream>>>(wq, wqT, DM, DM);
  cvt_transpose_f16<<<(nW + 255) / 256, 256, 0, stream>>>(wk, wkT, DM, DM);
  cvt_transpose_f16<<<(nW + 255) / 256, 256, 0, stream>>>(wv, wvT, DM, DM);
  cvt_f32_f16<<<(nW + 255) / 256, 256, 0, stream>>>(wo, woB, nW);
  const int nF = DM * FFN_DIM;
  cvt_transpose_f16<<<(nF + 255) / 256, 256, 0, stream>>>(w1, w1T, DM, FFN_DIM);
  cvt_transpose_f16<<<(nF + 255) / 256, 256, 0, stream>>>(w2, w2T, FFN_DIM, DM);

  // xn = rms_norm(x)
  rmsnorm_kernel<<<BS, 256, 0, stream>>>(x, xn, DM);

  const dim3 blk(128);
  const dim3 gQ(DM / 128, BS / 64);
  gemm_f16w<0><<<gQ, blk, 0, stream>>>(xn, wqT, BS, DM, DM, nullptr, nullptr, qb, nullptr);
  gemm_f16w<0><<<gQ, blk, 0, stream>>>(xn, wkT, BS, DM, DM, nullptr, nullptr, kbuf, nullptr);
  gemm_f16w<4><<<gQ, blk, 0, stream>>>(xn, wvT, BS, DM, DM, nullptr, nullptr, vbufT, nullptr);

  const dim3 gA(SEQ / 64, HEADS, 2);
  attn_kernel<<<gA, blk, 0, stream>>>(qb, kbuf, vbufT, attn);

  // x2 = x + attn @ w_o^T
  gemm_f16w<2><<<gQ, blk, 0, stream>>>(attn, woB, BS, DM, DM, nullptr, x, nullptr, x2);

  // xn = rms_norm(x2)
  rmsnorm_kernel<<<BS, 256, 0, stream>>>(x2, xn, DM);

  // h = relu6(xn @ w1 + b1)
  const dim3 gF1(FFN_DIM / 128, BS / 64);
  gemm_f16w<1><<<gF1, blk, 0, stream>>>(xn, w1T, BS, FFN_DIM, DM, b1, nullptr, hbuf, nullptr);

  // out = x2 + h @ w2 + b2
  const dim3 gF2(DM / 128, BS / 64);
  gemm_f16w<3><<<gF2, blk, 0, stream>>>(hbuf, w2T, BS, DM, FFN_DIM, b2, x2, nullptr, out);
}
